// Encoder_50629074485982
// MI455X (gfx1250) — matrix operand order repaired and checked by reading, not compiled, not run on hardware
//
#include <hip/hip_runtime.h>

// MI455X (gfx1250) LSTM encoder.
// Per-(t,layer) fused cell kernel: z = x@Wx + h@Wh + b via v_wmma_f32_16x16x32_f16.
// Each wave: 2 M-tiles x 4 gate tiles (8 WMMA / k-tile); BOTH A and B fragments
// double-buffered one k-step ahead, B prefetched two k-steps ahead (WGP scope).
// Weights are L2-resident (~40MB < 192MB L2); A (h state, 256KB) is WGP$-resident.

typedef __attribute__((ext_vector_type(16))) _Float16 v16h;
typedef __attribute__((ext_vector_type(8)))  _Float16 h8;
typedef __attribute__((ext_vector_type(8)))  float    v8f;

#define B_   128
#define T_   512
#define E_   64
#define D_   1024
#define ND4  4096   // 4*D

__device__ __forceinline__ float sigf(float x) { return 1.0f / (1.0f + __expf(-x)); }

// A fragment (16x32 f16) for wave32 WMMA from row-major [rows, ld] f16 matrix.
// lanes 0-15: M=lane, halves 0..7 -> K=kb..+7, halves 8..15 -> K=kb+16..+23
// lanes 16-31: M=lane-16, halves 0..7 -> K=kb+8..+15, halves 8..15 -> K=kb+24..+31
__device__ __forceinline__ v16h loadA(const _Float16* __restrict__ base, long ld,
                                      int m0, int kb, int lane) {
  int m  = m0 + (lane & 15);
  int kg = lane >> 4;
  const _Float16* p = base + (long)m * ld + kb + kg * 8;
  h8 lo = *(const h8*)p;
  h8 hi = *(const h8*)(p + 16);
  return __builtin_shufflevector(lo, hi, 0,1,2,3,4,5,6,7,8,9,10,11,12,13,14,15);
}

__device__ __forceinline__ v16h ld16(const _Float16* __restrict__ p) {
  h8 lo = *(const h8*)p;
  h8 hi = *(const h8*)(p + 8);
  return __builtin_shufflevector(lo, hi, 0,1,2,3,4,5,6,7,8,9,10,11,12,13,14,15);
}

// B fragments for the 4 gates of one (kt, db): contiguous 4KB block,
// layout [kt][db][gate][lane][16] halves.
struct Frag4 { v16h i, f, g, o; };
__device__ __forceinline__ Frag4 loadB4(const _Float16* __restrict__ frag,
                                        int kt, int db, int lane) {
  const _Float16* p = frag + (((long)((kt << 6) + db)) << 11) + (lane << 4);
  Frag4 r;
  r.i = ld16(p);
  r.f = ld16(p + 512);
  r.g = ld16(p + 1024);
  r.o = ld16(p + 1536);
  return r;
}

#define WMMA(acc, a, b) \
  acc = __builtin_amdgcn_wmma_f32_16x16x32_f16(false, (a), false, (b), (short)0, (acc), false, false)

// One LSTM cell step: z = xin@Wx + hprev@Wh + bias; gates fused in epilogue.
// Grid: 128 blocks x 64 threads (2 waves). db = blockIdx&63; each wave owns
// M-tiles {m0, m0+16} and the 4 gate tiles of d-block db.
__global__ void __launch_bounds__(64)
lstm_cell_kernel(const _Float16* __restrict__ xin, long xld, int ktx,
                 const _Float16* __restrict__ fragX,
                 const _Float16* __restrict__ hprev,
                 const _Float16* __restrict__ fragH,
                 const float* __restrict__ bias,
                 float* __restrict__ c,            // in-place c state [B,D]
                 _Float16* __restrict__ hnew) {    // [B,D] f16
  const int lane = threadIdx.x & 31;
  const int db   = blockIdx.x & 63;
  const int m0   = (((blockIdx.x >> 6) << 1) + (threadIdx.x >> 5)) << 5; // 0,32,64,96

  v8f Ai[2] = {{}, {}}, Af[2] = {{}, {}}, Ag[2] = {{}, {}}, Ao[2] = {{}, {}};

  // ---- part 1: input projection (K = ktx*32) ----
  {
    Frag4 b  = loadB4(fragX, 0, db, lane);
    v16h a0  = loadA(xin, xld, m0,      0, lane);
    v16h a1  = loadA(xin, xld, m0 + 16, 0, lane);
    for (int kt = 0; kt < ktx; ++kt) {
      int nkt = (kt + 1 < ktx) ? kt + 1 : kt;
      if (kt + 2 < ktx)   // pull kt+2's 4KB block into near caches (WGP$)
        __builtin_prefetch(fragX + (((long)(((kt + 2) << 6) + db)) << 11) + (lane << 7), 0, 3);
      Frag4 nb = loadB4(fragX, nkt, db, lane);
      v16h na0 = loadA(xin, xld, m0,      nkt * 32, lane);
      v16h na1 = loadA(xin, xld, m0 + 16, nkt * 32, lane);
      WMMA(Ai[0], a0, b.i); WMMA(Ai[1], a1, b.i);
      WMMA(Af[0], a0, b.f); WMMA(Af[1], a1, b.f);
      WMMA(Ag[0], a0, b.g); WMMA(Ag[1], a1, b.g);
      WMMA(Ao[0], a0, b.o); WMMA(Ao[1], a1, b.o);
      b = nb; a0 = na0; a1 = na1;
    }
  }
  // ---- part 2: recurrent projection (K = D) ----
  {
    Frag4 b  = loadB4(fragH, 0, db, lane);
    v16h a0  = loadA(hprev, D_, m0,      0, lane);
    v16h a1  = loadA(hprev, D_, m0 + 16, 0, lane);
    for (int kt = 0; kt < D_ / 32; ++kt) {
      int nkt = (kt + 1 < D_ / 32) ? kt + 1 : kt;
      if (kt + 2 < D_ / 32)
        __builtin_prefetch(fragH + (((long)(((kt + 2) << 6) + db)) << 11) + (lane << 7), 0, 3);
      Frag4 nb = loadB4(fragH, nkt, db, lane);
      v16h na0 = loadA(hprev, D_, m0,      nkt * 32, lane);
      v16h na1 = loadA(hprev, D_, m0 + 16, nkt * 32, lane);
      WMMA(Ai[0], a0, b.i); WMMA(Ai[1], a1, b.i);
      WMMA(Af[0], a0, b.f); WMMA(Af[1], a1, b.f);
      WMMA(Ag[0], a0, b.g); WMMA(Ag[1], a1, b.g);
      WMMA(Ao[0], a0, b.o); WMMA(Ao[1], a1, b.o);
      b = nb; a0 = na0; a1 = na1;
    }
  }

  // fused gate epilogue; C layout: lane holds (m = r + 8*(lane>>4), n = lane&15)
  const int d  = (db << 4) + (lane & 15);
  const float bi = bias[d], bf = bias[D_ + d], bg = bias[2 * D_ + d], bo = bias[3 * D_ + d];
  #pragma unroll
  for (int t2 = 0; t2 < 2; ++t2) {
    const int mb = m0 + (t2 << 4) + ((lane >> 4) << 3);
    #pragma unroll
    for (int r = 0; r < 8; ++r) {
      long idx = (long)(mb + r) * D_ + d;
      float gi = sigf(Ai[t2][r] + bi);
      float gf = sigf(Af[t2][r] + bf);
      float gg = tanhf(Ag[t2][r] + bg);
      float go = sigf(Ao[t2][r] + bo);
      float cn = gf * c[idx] + gi * gg;
      c[idx] = cn;
      hnew[idx] = (_Float16)(go * tanhf(cn));
    }
  }
}

// Pack fp32 weight [K, 4096] into f16 WMMA B-fragments, gate-adjacent order:
// frag layout [kt][db][gate][lane][j]; element (k,n):
//   kt=k/32, db=(n%1024)/16, gate=n/1024, lane=(n%16)+16*((k%16)/8), j=(k%8)+8*((k%32)/16)
__global__ void pack_w_kernel(const float* __restrict__ W, _Float16* __restrict__ frag) {
  long idx = (long)blockIdx.x * blockDim.x + threadIdx.x;
  int j    = idx & 15;
  int lane = (idx >> 4) & 31;
  int gate = (idx >> 9) & 3;
  int db   = (idx >> 11) & 63;
  int kt   = (int)(idx >> 17);
  int k = kt * 32 + (lane >> 4) * 8 + (j & 7) + (j >> 3) * 16;
  int n = gate * D_ + db * 16 + (lane & 15);
  frag[idx] = (_Float16)W[(long)k * ND4 + n];
}

__global__ void f32_to_f16_kernel(const float* __restrict__ src,
                                  _Float16* __restrict__ dst, long n) {
  long i = (long)blockIdx.x * blockDim.x + threadIdx.x;
  if (i < n) dst[i] = (_Float16)src[i];
}

__global__ void zero_f16_kernel(_Float16* __restrict__ p, long n) {
  long i = (long)blockIdx.x * blockDim.x + threadIdx.x;
  if (i < n) p[i] = (_Float16)0.0f;
}

__global__ void zero_f32_kernel(float* __restrict__ p, long n) {
  long i = (long)blockIdx.x * blockDim.x + threadIdx.x;
  if (i < n) p[i] = 0.0f;
}

// VAE head: out = feat@w_mean + b_mean + exp((feat@w_sigma + b_sigma)/2) * eps
__global__ void __launch_bounds__(256)
head_kernel(const float* __restrict__ feat,          // [128,1024] (last-layer c)
            const float* __restrict__ wm, const float* __restrict__ bm,
            const float* __restrict__ wsg, const float* __restrict__ bs,
            const float* __restrict__ eps, float* __restrict__ out) {
  int tid = blockIdx.x * blockDim.x + threadIdx.x;  // 0..16383
  int b = tid >> 7;
  int j = tid & 127;
  float sm = bm[j], ss = bs[j];
  const float* fr = feat + (long)b * D_;
  #pragma unroll 4
  for (int k = 0; k < D_; ++k) {
    float f = fr[k];
    sm += f * wm[(long)k * 128 + j];
    ss += f * wsg[(long)k * 128 + j];
  }
  out[tid] = sm + __expf(0.5f * ss) * eps[tid];
}

extern "C" void kernel_launch(void* const* d_in, const int* in_sizes, int n_in,
                              void* d_out, int out_size, void* d_ws, size_t ws_size,
                              hipStream_t stream) {
  (void)in_sizes; (void)n_in; (void)out_size; (void)ws_size;
  const float* inputs  = (const float*)d_in[0];
  const float* k0      = (const float*)d_in[1];
  const float* rk0     = (const float*)d_in[2];
  const float* b0      = (const float*)d_in[3];
  const float* k1      = (const float*)d_in[4];
  const float* rk1     = (const float*)d_in[5];
  const float* b1      = (const float*)d_in[6];
  const float* k2      = (const float*)d_in[7];
  const float* rk2     = (const float*)d_in[8];
  const float* b2      = (const float*)d_in[9];
  const float* w_mean  = (const float*)d_in[10];
  const float* b_mean  = (const float*)d_in[11];
  const float* w_sigma = (const float*)d_in[12];
  const float* b_sigma = (const float*)d_in[13];
  const float* eps     = (const float*)d_in[14];
  float* out = (float*)d_out;

  // Workspace carve-up (256B aligned)
  size_t off = 0;
  char* base = (char*)d_ws;
  auto carve = [&](size_t bytes) -> void* {
    void* p = base + off;
    off += (bytes + 255) & ~(size_t)255;
    return p;
  };
  const size_t fragBytes = (size_t)32 << 18;                      // K=1024 -> 8MB
  _Float16* f0x = (_Float16*)carve((size_t)2 << 18);              // K=64 -> 0.5MB
  _Float16* f0h = (_Float16*)carve(fragBytes);
  _Float16* f1x = (_Float16*)carve(fragBytes);
  _Float16* f1h = (_Float16*)carve(fragBytes);
  _Float16* f2x = (_Float16*)carve(fragBytes);
  _Float16* f2h = (_Float16*)carve(fragBytes);
  _Float16* xf16 = (_Float16*)carve((size_t)B_ * T_ * E_ * sizeof(_Float16));
  _Float16* hb[3][2];
  for (int l = 0; l < 3; ++l)
    for (int p = 0; p < 2; ++p)
      hb[l][p] = (_Float16*)carve((size_t)B_ * D_ * sizeof(_Float16));
  float* cb[3];
  for (int l = 0; l < 3; ++l)
    cb[l] = (float*)carve((size_t)B_ * D_ * sizeof(float));

  // ---- one-time prep (deterministic, rerun every call) ----
  pack_w_kernel<<<2  * 512, 256, 0, stream>>>(k0,  f0x);
  pack_w_kernel<<<32 * 512, 256, 0, stream>>>(rk0, f0h);
  pack_w_kernel<<<32 * 512, 256, 0, stream>>>(k1,  f1x);
  pack_w_kernel<<<32 * 512, 256, 0, stream>>>(rk1, f1h);
  pack_w_kernel<<<32 * 512, 256, 0, stream>>>(k2,  f2x);
  pack_w_kernel<<<32 * 512, 256, 0, stream>>>(rk2, f2h);
  {
    long n = (long)B_ * T_ * E_;
    f32_to_f16_kernel<<<(unsigned)((n + 255) / 256), 256, 0, stream>>>(inputs, xf16, n);
  }
  {
    long nh = (long)B_ * D_;
    for (int l = 0; l < 3; ++l) {
      zero_f16_kernel<<<(unsigned)((nh + 255) / 256), 256, 0, stream>>>(hb[l][0], nh);
      zero_f16_kernel<<<(unsigned)((nh + 255) / 256), 256, 0, stream>>>(hb[l][1], nh);
      zero_f32_kernel<<<(unsigned)((nh + 255) / 256), 256, 0, stream>>>(cb[l], nh);
    }
  }

  // ---- sequential recurrence: 3 fused cell GEMMs per timestep ----
  for (int t = 0; t < T_; ++t) {
    int p = t & 1, q = p ^ 1;
    // layer 0: x from converted inputs, K_x = 64 (2 k-tiles)
    lstm_cell_kernel<<<128, 64, 0, stream>>>(
        xf16 + (long)t * E_, (long)T_ * E_, 2, f0x,
        hb[0][p], f0h, b0, cb[0], hb[0][q]);
    // layer 1: x = h of layer 0 (this step), K_x = 1024
    lstm_cell_kernel<<<128, 64, 0, stream>>>(
        hb[0][q], (long)D_, 32, f1x,
        hb[1][p], f1h, b1, cb[1], hb[1][q]);
    // layer 2
    lstm_cell_kernel<<<128, 64, 0, stream>>>(
        hb[1][q], (long)D_, 32, f2x,
        hb[2][p], f2h, b2, cb[2], hb[2][q]);
  }

  // ---- VAE head on last-layer CELL state (reference uses cs[-1]) ----
  head_kernel<<<64, 256, 0, stream>>>(cb[2], w_mean, b_mean, w_sigma, b_sigma, eps, out);
}
